// MLP_mia_65300682768662
// MI455X (gfx1250) — matrix operand order repaired and checked by reading, not compiled, not run on hardware
//
#include <hip/hip_runtime.h>

#define CDNA5_EPS 1e-8f
#define EMB_D 256

typedef __attribute__((ext_vector_type(16))) _Float16 v16h;
typedef __attribute__((ext_vector_type(8)))  _Float16 v8h;
typedef __attribute__((ext_vector_type(8)))  float    v8f;
typedef __attribute__((ext_vector_type(4)))  float    v4f;

__device__ __forceinline__ v16h zero_v16h() {
  v16h v;
#pragma unroll
  for (int i = 0; i < 16; ++i) v[i] = (_Float16)0.0f;
  return v;
}

__device__ __forceinline__ v8f splat_v8f(float x) {
  v8f v;
#pragma unroll
  for (int i = 0; i < 8; ++i) v[i] = x;
  return v;
}

__device__ __forceinline__ float wave_red_add(float v) {
#pragma unroll
  for (int m = 16; m >= 1; m >>= 1) v += __shfl_xor(v, m, 32);
  return v;
}

// One wave32 processes a tile of 16 edges end-to-end:
//   gather + weighted cos/dot (VALU, b128 loads) -> tiny MLP via v_wmma_f32_16x16x32_f16.
__global__ __launch_bounds__(256)
void edge_cos_mlp_kernel(const int* __restrict__ edge,
                         const float* __restrict__ emb,
                         const float* __restrict__ wvec,
                         const float* __restrict__ W1, const float* __restrict__ b1,
                         const float* __restrict__ W2, const float* __restrict__ b2,
                         const float* __restrict__ Wp, const float* __restrict__ bp,
                         float* __restrict__ out, int E)
{
  constexpr int STRIDE = 72; // f16 elems per LDS row (144B, 16B-aligned runs)
  __shared__ _Float16 ldsH[8][16 * STRIDE];

  const int lane   = threadIdx.x & 31;
  const int waveId = threadIdx.x >> 5;
  const int tile   = blockIdx.x * 8 + waveId;
  const int numTiles = (E + 15) >> 4;
  if (tile >= numTiles) return;          // wave-uniform exit; survivors have full EXEC
  const int base = tile << 4;

  const int lc   = lane & 15;
  const int half = lane >> 4;

  // Per-lane 8-dim slice of the learned weight vector (loaded once).
  const v4f wA = *(const v4f*)(wvec + lane * 8);
  const v4f wB = *(const v4f*)(wvec + lane * 8 + 4);

  // ---- Stage 1: gather rows, weighted cos/dot per edge (gather-bound part) ----
  float cosv = 0.f, dotv = 0.f;
  for (int i = 0; i < 16; ++i) {
    const int e = base + i;              // uniform across wave
    if (e < E) {
      const int i0 = edge[2 * e + 0];
      const int i1 = edge[2 * e + 1];
      const float* r1 = emb + (size_t)i0 * EMB_D + lane * 8;
      const float* r2 = emb + (size_t)i1 * EMB_D + lane * 8;
      v4f a0 = *(const v4f*)(r1);
      v4f a1 = *(const v4f*)(r1 + 4);
      v4f c0 = *(const v4f*)(r2);
      v4f c1 = *(const v4f*)(r2 + 4);
      v4f x0 = a0 * wA, x1 = a1 * wB;    // x1 row, weighted
      v4f y0 = c0 * wA, y1 = c1 * wB;    // x2 row, weighted
      float pd = 0.f, pn1 = 0.f, pn2 = 0.f;
#pragma unroll
      for (int k = 0; k < 4; ++k) {
        pd  += x0[k] * y0[k] + x1[k] * y1[k];
        pn1 += x0[k] * x0[k] + x1[k] * x1[k];
        pn2 += y0[k] * y0[k] + y1[k] * y1[k];
      }
      const float dsum = wave_red_add(pd);
      const float n1s  = wave_red_add(pn1);
      const float n2s  = wave_red_add(pn2);
      if (lane == i) {                   // edge i's (cos, dot) lives on lane i
        const float n1 = fmaxf(sqrtf(n1s), CDNA5_EPS);
        const float n2 = fmaxf(sqrtf(n2s), CDNA5_EPS);
        dotv = dsum;
        cosv = dsum / (n1 * n2);
      }
    }
  }

  // ---- Stage 2: layer 1, interaction[16,2] @ W1^T[2,64] + b1, via WMMA ----
  // A (16x32 f16): lanes 0-15 = rows, elements 0,1 = K0,K1 (cos,dot); rest zero.
  v16h A1 = zero_v16h();
  if (half == 0) {
    A1[0] = (_Float16)cosv;
    A1[1] = (_Float16)dotv;
  }

  v8f h1[4];
#pragma unroll
  for (int j = 0; j < 4; ++j) {          // N = 64 -> four 16-wide tiles
    v16h B1 = zero_v16h();
    if (half == 0) {                     // B: lane = column, elements 0,1 = K0,K1
      const int row = 16 * j + lc;       // W1 row = output neuron
      B1[0] = (_Float16)W1[row * 2 + 0];
      B1[1] = (_Float16)W1[row * 2 + 1];
    }
    v8f c = splat_v8f(b1[16 * j + lc]);  // bias in C (D-layout: column = lane%16)
    h1[j] = __builtin_amdgcn_wmma_f32_16x16x32_f16(false, A1, false, B1,
                                                   (short)0, c, false, false);
  }

  // ---- Stage 3: relu(h1) -> LDS (f16, row-major) -> reload as A-operands ----
  _Float16* myLds = &ldsH[waveId][0];
#pragma unroll
  for (int j = 0; j < 4; ++j) {
#pragma unroll
    for (int r = 0; r < 8; ++r) {
      const int row = r + 8 * half;      // D-layout: VGPR r -> row r (+8 for hi half)
      const int col = 16 * j + lc;
      myLds[row * STRIDE + col] = (_Float16)fmaxf(h1[j][r], 0.f);
    }
  }
  asm volatile("s_wait_dscnt 0" ::: "memory"); // same-wave DS is in-order; be explicit

  // A reload (contiguous k, same as B2): lane m<16 takes K {0..15}+32t of row m; lanes>=16: {16..31}.
  v16h A2[2];
#pragma unroll
  for (int t = 0; t < 2; ++t) {
    const int kb = 32 * t + 16 * half;
    const v8h lo = *(const v8h*)(myLds + lc * STRIDE + kb);        // ds_load_b128
    const v8h hi = *(const v8h*)(myLds + lc * STRIDE + kb + 8);    // ds_load_b128
    A2[t] = __builtin_shufflevector(lo, hi, 0,1,2,3,4,5,6,7,8,9,10,11,12,13,14,15);
  }

  // ---- Stage 4: layer 2, h1[16,64] @ W2^T[64,32] + b2, via WMMA (K split 2x32) ----
  // B tile [n][t]: lane j holds W2[16n+j][32t + 16*half + 0..15] (contiguous b128 loads).
  v16h B2[2][2];
#pragma unroll
  for (int n = 0; n < 2; ++n) {
#pragma unroll
    for (int t = 0; t < 2; ++t) {
      const float* p = W2 + (size_t)(16 * n + lc) * 64 + 32 * t + 16 * half;
      v16h b;
#pragma unroll
      for (int q = 0; q < 4; ++q) {
        const v4f f = *(const v4f*)(p + 4 * q);
#pragma unroll
        for (int k = 0; k < 4; ++k) b[4 * q + k] = (_Float16)f[k];
      }
      B2[n][t] = b;
    }
  }

  v8f h2[2];
#pragma unroll
  for (int n = 0; n < 2; ++n) {
    v8f c = splat_v8f(b2[16 * n + lc]);
    c = __builtin_amdgcn_wmma_f32_16x16x32_f16(false, A2[0], false, B2[n][0],
                                               (short)0, c, false, false);
    c = __builtin_amdgcn_wmma_f32_16x16x32_f16(false, A2[1], false, B2[n][1],
                                               (short)0, c, false, false);
    h2[n] = c;
  }

  // ---- Stage 5: pred = relu(h2) . Wp + bp, reduced across the 16-lane halves ----
  const float wlo = Wp[lc];
  const float whi = Wp[16 + lc];
  const float bpv = bp[0];

  float acc[8];
#pragma unroll
  for (int r = 0; r < 8; ++r)
    acc[r] = fmaxf(h2[0][r], 0.f) * wlo + fmaxf(h2[1][r], 0.f) * whi;

#pragma unroll
  for (int m = 8; m >= 1; m >>= 1) {
#pragma unroll
    for (int r = 0; r < 8; ++r)
      acc[r] += __shfl_xor(acc[r], m, 32);   // masks <16: stays within each half
  }

  if (lc < 8) {                              // lanes 0-7 -> rows 0-7; 16-23 -> rows 8-15
    float v = acc[0];
#pragma unroll
    for (int r = 1; r < 8; ++r) v = (lc == r) ? acc[r] : v;
    const int row = lc + 8 * half;
    const int e = base + row;
    if (e < E) out[e] = v + bpv;
  }
}

extern "C" void kernel_launch(void* const* d_in, const int* in_sizes, int n_in,
                              void* d_out, int out_size, void* d_ws, size_t ws_size,
                              hipStream_t stream) {
  const int*   edge = (const int*)  d_in[0]; // [E,2] indices
  const float* emb  = (const float*)d_in[1]; // [N,256]
  const float* wv   = (const float*)d_in[2]; // [1,256]
  const float* W1   = (const float*)d_in[3]; // [64,2]
  const float* b1   = (const float*)d_in[4]; // [64]
  const float* W2   = (const float*)d_in[5]; // [32,64]
  const float* b2   = (const float*)d_in[6]; // [32]
  const float* Wp   = (const float*)d_in[7]; // [1,32]
  const float* bp   = (const float*)d_in[8]; // [1]

  const int E = in_sizes[0] / 2;
  const int numTiles = (E + 15) / 16;
  const int blocks = (numTiles + 7) / 8;     // 8 waves (tiles) per 256-thread block
  if (blocks > 0) {
    edge_cos_mlp_kernel<<<blocks, 256, 0, stream>>>(
        edge, emb, wv, W1, b1, W2, b2, Wp, bp, (float*)d_out, E);
  }
}
